// lossEvaluator_23991687315906
// MI455X (gfx1250) — matrix operand order repaired and checked by reading, not compiled, not run on hardware
//
#include <hip/hip_runtime.h>
#include <hip/hip_bf16.h>

#define BB 64
#define PP 2048
#define DD 1024

typedef __attribute__((ext_vector_type(16))) __bf16 v16bf;
typedef __attribute__((ext_vector_type(8)))  __bf16 bf16x8;
typedef __attribute__((ext_vector_type(4)))  __bf16 bf16x4;
typedef __attribute__((ext_vector_type(4)))  float  f32x4;
typedef __attribute__((ext_vector_type(8)))  float  v8f;

// LDS row stride in bf16 elements: 32 data + 8 pad = 40 (80 B = 20 banks -> the
// 16 lanes of a fragment read hit 16 distinct 4-bank groups; 16B aligned).
#define LDSTRIDE 40

// float4 -> 4x bf16 via native packed converts (v_cvt_pk_bf16_f32), 8B LDS store
static __device__ __forceinline__ void cvt_store4(__bf16* dst, const float4 v) {
    f32x4 f = { v.x, v.y, v.z, v.w };
    *(bf16x4*)dst = __builtin_convertvector(f, bf16x4);
}

__global__ __launch_bounds__(256)
void simmax_kernel(const float* __restrict__ imFtr,
                   const float* __restrict__ disFtr,
                   float* __restrict__ partial /* [64][16][64] */) {
    __shared__ __bf16 ldsA[2][128 * LDSTRIDE];   // 2 x 10240 B
    __shared__ __bf16 ldsB[2][ 64 * LDSTRIDE];   // 2 x  5120 B
    __shared__ float  wmax[8 * 64];              //      2048 B

    const int tid  = threadIdx.x;
    const int wave = tid >> 5;
    const int lane = tid & 31;
    const int l16  = lane & 15;
    const int hf   = lane >> 4;         // which 16-lane half of the wave
    const int b    = blockIdx.y;
    const int rowTile = blockIdx.x;
    const size_t aBase = ((size_t)b * PP + (size_t)rowTile * 128) * DD;

    v8f acc[4] = {};                    // 16(M) x 64(N) per wave, f32 accum

    const int r8 = tid >> 3;            // 0..31
    const int c4 = (tid & 7) << 2;      // 0,4,...,28

    float4 pa[4], pb[2];                // register prefetch for next tile

    // ---- prologue: fetch + stage K-chunk 0 into buffer 0 ----
    #pragma unroll
    for (int rr = 0; rr < 4; ++rr)
        pa[rr] = *(const float4*)(imFtr + aBase + (size_t)(r8 + rr * 32) * DD + c4);
    #pragma unroll
    for (int rr = 0; rr < 2; ++rr)
        pb[rr] = *(const float4*)(disFtr + (size_t)(r8 + rr * 32) * DD + c4);
    #pragma unroll
    for (int rr = 0; rr < 4; ++rr)
        cvt_store4(&ldsA[0][(r8 + rr * 32) * LDSTRIDE + c4], pa[rr]);
    #pragma unroll
    for (int rr = 0; rr < 2; ++rr)
        cvt_store4(&ldsB[0][(r8 + rr * 32) * LDSTRIDE + c4], pb[rr]);
    __syncthreads();

    const int aRow = wave * 16 + l16;

    #pragma unroll 2
    for (int it = 0; it < DD / 32; ++it) {
        const int buf  = it & 1;
        const bool more = (it + 1) < (DD / 32);

        // ---- prefetch next K-chunk from global into registers (hidden by WMMA) ----
        if (more) {
            const int k0n = (it + 1) * 32;
            #pragma unroll
            for (int rr = 0; rr < 4; ++rr)
                pa[rr] = *(const float4*)(imFtr + aBase + (size_t)(r8 + rr * 32) * DD + k0n + c4);
            #pragma unroll
            for (int rr = 0; rr < 2; ++rr)
                pb[rr] = *(const float4*)(disFtr + (size_t)(r8 + rr * 32) * DD + k0n + c4);
        }

        // ---- fragments from current buffer ----
        // A (16x32 bf16): lane holds row (wave*16+l16), K = hf*8+{0..7} U hf*8+16+{0..7}
        bf16x8 alo = *(const bf16x8*)&ldsA[buf][aRow * LDSTRIDE + hf * 8];
        bf16x8 ahi = *(const bf16x8*)&ldsA[buf][aRow * LDSTRIDE + hf * 8 + 16];
        v16bf afrag = __builtin_shufflevector(alo, ahi,
            0,1,2,3,4,5,6,7,8,9,10,11,12,13,14,15);

        // B (32x16 bf16): lane holds col (nt*16+l16), K = hf*8+{0..7} U hf*8+16+{0..7}
        // Load all 4 fragments first so one dscnt wait covers the WMMA chain.
        bf16x8 blo[4], bhi[4];
        #pragma unroll
        for (int nt = 0; nt < 4; ++nt) {
            const int bRow = nt * 16 + l16;
            blo[nt] = *(const bf16x8*)&ldsB[buf][bRow * LDSTRIDE + hf * 8];
            bhi[nt] = *(const bf16x8*)&ldsB[buf][bRow * LDSTRIDE + hf * 8 + 16];
        }
        #pragma unroll
        for (int nt = 0; nt < 4; ++nt) {
            v16bf bfrag = __builtin_shufflevector(blo[nt], bhi[nt],
                0,1,2,3,4,5,6,7,8,9,10,11,12,13,14,15);
            acc[nt] = __builtin_amdgcn_wmma_f32_16x16x32_bf16(
                false, afrag, false, bfrag, (short)0, acc[nt], false, false);
        }

        // ---- stage prefetched tile into the other buffer ----
        if (more) {
            #pragma unroll
            for (int rr = 0; rr < 4; ++rr)
                cvt_store4(&ldsA[buf ^ 1][(r8 + rr * 32) * LDSTRIDE + c4], pa[rr]);
            #pragma unroll
            for (int rr = 0; rr < 2; ++rr)
                cvt_store4(&ldsB[buf ^ 1][(r8 + rr * 32) * LDSTRIDE + c4], pb[rr]);
        }
        __syncthreads();
    }

    // ---- max over this wave's 16 rows, per column ----
    // acc[nt][r] = sim(row = tile*128 + wave*16 + hf*8 + r, col = nt*16 + l16)
    #pragma unroll
    for (int nt = 0; nt < 4; ++nt) {
        float m = acc[nt][0];
        #pragma unroll
        for (int r = 1; r < 8; ++r) m = fmaxf(m, acc[nt][r]);
        m = fmaxf(m, __shfl_xor(m, 16, 32));   // merge the two row-halves
        if (hf == 0) wmax[wave * 64 + nt * 16 + l16] = m;
    }
    __syncthreads();

    // ---- merge 8 waves -> deterministic partial max for this (b, tile) ----
    if (tid < 64) {
        float m = wmax[tid];
        #pragma unroll
        for (int w = 1; w < 8; ++w) m = fmaxf(m, wmax[w * 64 + tid]);
        partial[((size_t)b * 16 + rowTile) * 64 + tid] = m;
    }
}

__global__ __launch_bounds__(256)
void loss_kernel(const float* __restrict__ partial,
                 const int* __restrict__ lbl,
                 float* __restrict__ out) {
    __shared__ float simMax[64 * 64];
    __shared__ float pos[64];
    __shared__ int   lblS[64];
    __shared__ float redH[256];
    __shared__ float redC[256];

    const int tid = threadIdx.x;

    // reduce the 16 row-tiles -> simMax[i][j]
    for (int idx = tid; idx < 4096; idx += 256) {
        const int i = idx >> 6, j = idx & 63;
        float m = partial[((size_t)i * 16 + 0) * 64 + j];
        #pragma unroll
        for (int t = 1; t < 16; ++t)
            m = fmaxf(m, partial[((size_t)i * 16 + t) * 64 + j]);
        simMax[idx] = m;
    }
    __syncthreads();
    if (tid < 64) { pos[tid] = simMax[tid * 64 + tid]; lblS[tid] = lbl[tid]; }
    __syncthreads();

    float sH = 0.f, sC = 0.f;
    for (int idx = tid; idx < 4096; idx += 256) {
        const int i = idx >> 6, j = idx & 63;
        if (lblS[i] != lblS[j]) {
            sC += 1.f;
            float h1 = simMax[i * 64 + j] - pos[i] + 0.1f;   // image -> text
            float h2 = simMax[j * 64 + i] - pos[i] + 0.1f;   // text  -> image
            sH += (h1 > 0.f ? h1 : 0.f) + (h2 > 0.f ? h2 : 0.f);
        }
    }
    redH[tid] = sH; redC[tid] = sC;
    __syncthreads();
    for (int s = 128; s > 0; s >>= 1) {
        if (tid < s) { redH[tid] += redH[tid + s]; redC[tid] += redC[tid + s]; }
        __syncthreads();
    }
    if (tid == 0) out[0] = redH[0] / (redC[0] + 1e-6f);
}

extern "C" void kernel_launch(void* const* d_in, const int* in_sizes, int n_in,
                              void* d_out, int out_size, void* d_ws, size_t ws_size,
                              hipStream_t stream) {
    const float* imFtr  = (const float*)d_in[0];
    const float* disFtr = (const float*)d_in[1];
    const int*   lbl    = (const int*)d_in[2];
    float* out = (float*)d_out;
    float* partial = (float*)d_ws;   // needs 64*16*64*4 = 256 KB

    dim3 grid(16, 64);               // (row tiles of 128, batch b)
    simmax_kernel<<<grid, 256, 0, stream>>>(imFtr, disFtr, partial);
    loss_kernel<<<1, 256, 0, stream>>>(partial, lbl, out);
}
